// Model_39676907882095
// MI455X (gfx1250) — matrix operand order repaired and checked by reading, not compiled, not run on hardware
//
#include <hip/hip_runtime.h>

// Problem constants (reference: B=4, S=2048, D=512, scale = 1/sqrt(64) = 0.125)
#define BATCH 4
#define SEQ   2048
#define DIM   512
#define SCALEF 0.125f

// Block tile: 64(M) x 128(N) x 32(K) per 256-thread block (8 waves of 32).
// Waves arranged 2x4; each wave owns a 32x32 macro-tile:
//   2 A frags x 2 B frags -> 4 v_wmma_f32_16x16x32_bf16 per K-step.
// LDS tiles are double-buffered; async global->LDS DMA for tile i+1 is issued
// before computing tile i, then s_wait_asynccnt <= (ops just issued) proves
// (by in-order ASYNCcnt completion) that tile i has fully landed.
#define BM 64
#define BN 128
#define BK 32

typedef __attribute__((ext_vector_type(16))) __bf16 v16bf;
typedef __attribute__((ext_vector_type(8)))  __bf16 v8bf;
typedef __attribute__((ext_vector_type(8)))  float  v8f;

// ---------------------------------------------------------------------------
// CDNA5 async global->LDS copy (ASYNCcnt-tracked DMA, 16B per lane) + waits.
// LDS dest is an absolute LDS byte offset; we use only dynamic LDS (no static
// __shared__), so the dynamic region starts at LDS offset 0.
// ---------------------------------------------------------------------------
__device__ __forceinline__ void async_g2l_b128(unsigned lds_byte_off,
                                               const void* gaddr) {
  unsigned long long ga = (unsigned long long)gaddr;
  asm volatile("global_load_async_to_lds_b128 %0, %1, off"
               :: "v"(lds_byte_off), "v"(ga) : "memory");
}
__device__ __forceinline__ void wait_async_le0() { asm volatile("s_wait_asynccnt 0x0" ::: "memory"); }
__device__ __forceinline__ void wait_async_le2() { asm volatile("s_wait_asynccnt 0x2" ::: "memory"); }
__device__ __forceinline__ void wait_async_le6() { asm volatile("s_wait_asynccnt 0x6" ::: "memory"); }

// ---- WMMA fragment builders (contiguous k order, same for A and B, wave32) ----
// A 16-bit 16x32: lane L(0-15)=row M, elements hold K 0..15;
// lanes 16-31 same rows with K 16..31.
__device__ __forceinline__ v16bf frag_a_from_f32(const float* __restrict__ rowbase, int half) {
  v16bf a;
#pragma unroll
  for (int e = 0; e < 16; ++e) {
    const int K = e + half * 16;
    a[e] = (__bf16)rowbase[K];
  }
  return a;
}
__device__ __forceinline__ v16bf frag_b_from_f32(const float* __restrict__ rowbase, int half) {
  v16bf b;
#pragma unroll
  for (int e = 0; e < 16; ++e) {
    const int K = e + half * 16;
    b[e] = (__bf16)rowbase[K];
  }
  return b;
}
__device__ __forceinline__ v16bf frag_a_from_bf16(const __bf16* __restrict__ rowbase, int half) {
  v16bf a;
#pragma unroll
  for (int e = 0; e < 16; ++e) {
    const int K = e + half * 16;
    a[e] = rowbase[K];
  }
  return a;
}
// B fragment gathered from K-major LDS (Bsk[K][n]): lane L(0-15)=col N with
// K 0..15, lanes 16-31 same cols with K 16..31 (same-dword u16 reads multicast).
__device__ __forceinline__ v16bf frag_b_from_kmajor(const __bf16* __restrict__ base,
                                                    int n, int pitch, int half) {
  v16bf b;
#pragma unroll
  for (int e = 0; e < 16; ++e) {
    const int K = e + half * 16;
    b[e] = base[K * pitch + n];
  }
  return b;
}

#define WMMA_BF16(A, Bf, C) \
  __builtin_amdgcn_wmma_f32_16x16x32_bf16(false, (A), false, (Bf), (short)0, (C), false, false)

// ============================================================================
// Kernel 1: qk = q @ k^T  (f32 in via async DMA, bf16 out to workspace)
// q[s,d] and k[t,d] are both K(=d)-contiguous -> async b128 copies land
// fragment-ready (no transpose). f32->bf16 conversion folds into frag loads
// (v_cvt_pk_bf16_f32 co-executes with WMMA).
// ============================================================================
#define P1 36                               // f32 elems per LDS row (144B, 16B-aligned)
#define A1_OFF 0
#define B1_OFF (BM * P1 * 4)                // 9216
#define LDS1_HALF (B1_OFF + BN * P1 * 4)    // 27648 per buffer
#define LDS1_BYTES (2 * LDS1_HALF)          // 55296

__device__ __forceinline__ void stage_qk(const float* __restrict__ qb,
                                         const float* __restrict__ kb,
                                         int s0, int n0, int k0,
                                         unsigned base, int tid) {
#pragma unroll
  for (int it = 0; it < 2; ++it) {          // A: 64 rows x 8 float4 chunks
    const int c = tid + it * 256;
    const int row = c >> 3, c4 = c & 7;
    async_g2l_b128(base + A1_OFF + row * (P1 * 4) + c4 * 16,
                   qb + (size_t)(s0 + row) * DIM + k0 + c4 * 4);
  }
#pragma unroll
  for (int it = 0; it < 4; ++it) {          // B: 128 rows x 8 float4 chunks
    const int c = tid + it * 256;
    const int row = c >> 3, c4 = c & 7;
    async_g2l_b128(base + B1_OFF + row * (P1 * 4) + c4 * 16,
                   kb + (size_t)(n0 + row) * DIM + k0 + c4 * 4);
  }
}

__global__ __launch_bounds__(256) void qk_gemm_kernel(const float* __restrict__ q,
                                                      const float* __restrict__ k,
                                                      __bf16* __restrict__ qkb) {
  extern __shared__ char smem[];

  const int b    = blockIdx.z;
  const int n0   = blockIdx.x * BN;
  const int s0   = blockIdx.y * BM;
  const int tid  = threadIdx.x;
  const int lane = tid & 31;
  const int wave = tid >> 5;
  const int wr   = wave >> 2;          // wave row 0..1 (32 rows each)
  const int wc   = wave & 3;           // wave col 0..3 (32 cols each)
  const int lr   = lane & 15;
  const int half = lane >> 4;

  const float* __restrict__ qb = q + (size_t)b * SEQ * DIM;
  const float* __restrict__ kb = k + (size_t)b * SEQ * DIM;

  v8f c00 = {}, c01 = {}, c10 = {}, c11 = {};

  const int nsteps = DIM / BK;         // 16
  stage_qk(qb, kb, s0, n0, 0, 0u, tid);           // prologue: tile 0 -> buf 0

  for (int step = 0; step < nsteps; ++step) {
    const unsigned cur = (unsigned)(step & 1) * LDS1_HALF;
    if (step + 1 < nsteps) {
      __syncthreads();                 // all waves done reading the other buffer
      stage_qk(qb, kb, s0, n0, (step + 1) * BK, cur ^ LDS1_HALF, tid);
      wait_async_le6();                // tile `step` (6 older ops) has landed
    } else {
      wait_async_le0();
    }
    __syncthreads();                   // everyone's tile `step` visible

    const float* Af = (const float*)(smem + cur + A1_OFF);
    const float* Bf = (const float*)(smem + cur + B1_OFF);
    const v16bf a0 = frag_a_from_f32(Af + (wr * 32 + lr) * P1, half);
    const v16bf a1 = frag_a_from_f32(Af + (wr * 32 + 16 + lr) * P1, half);
    const v16bf b0 = frag_b_from_f32(Bf + (wc * 32 + lr) * P1, half);
    const v16bf b1 = frag_b_from_f32(Bf + (wc * 32 + 16 + lr) * P1, half);
    c00 = WMMA_BF16(a0, b0, c00);
    c01 = WMMA_BF16(a0, b1, c01);
    c10 = WMMA_BF16(a1, b0, c10);
    c11 = WMMA_BF16(a1, b1, c11);
  }

  // C/D layout: vgpr jj holds M = jj + 8*half, N = lane%16
  __bf16* __restrict__ ob = qkb + (size_t)b * SEQ * SEQ;
#pragma unroll
  for (int jj = 0; jj < 8; ++jj) {
    const int r0 = s0 + wr * 32 + jj + half * 8;
    const int cc = n0 + wc * 32 + lr;
    ob[(size_t)r0 * SEQ + cc]             = (__bf16)c00[jj];
    ob[(size_t)r0 * SEQ + cc + 16]        = (__bf16)c01[jj];
    ob[(size_t)(r0 + 16) * SEQ + cc]      = (__bf16)c10[jj];
    ob[(size_t)(r0 + 16) * SEQ + cc + 16] = (__bf16)c11[jj];
  }
}

// ============================================================================
// Kernel 2: per-row softmax stats over scores = qk * SCALE (max and sum-exp)
// ============================================================================
__global__ __launch_bounds__(256) void softmax_stats_kernel(const __bf16* __restrict__ qkb,
                                                            float* __restrict__ mrow,
                                                            float* __restrict__ lrow) {
  __shared__ float red[256];
  const int row = blockIdx.x;
  const __bf16* __restrict__ p = qkb + (size_t)row * SEQ;
  const int tid = threadIdx.x;

  float mx = -3.0e38f;
#pragma unroll
  for (int it = 0; it < SEQ / 256; ++it)
    mx = fmaxf(mx, (float)p[tid + it * 256] * SCALEF);
  red[tid] = mx;
  __syncthreads();
  for (int s = 128; s > 0; s >>= 1) {
    if (tid < s) red[tid] = fmaxf(red[tid], red[tid + s]);
    __syncthreads();
  }
  mx = red[0];
  __syncthreads();

  float sum = 0.0f;
#pragma unroll
  for (int it = 0; it < SEQ / 256; ++it)
    sum += __expf((float)p[tid + it * 256] * SCALEF - mx);
  red[tid] = sum;
  __syncthreads();
  for (int s = 128; s > 0; s >>= 1) {
    if (tid < s) red[tid] += red[tid + s];
    __syncthreads();
  }
  if (tid == 0) { mrow[row] = mx; lrow[row] = red[0]; }
}

// ============================================================================
// Kernel 3: out = softmax(qk*SCALE) @ qk   (K = SEQ = 2048)
// B tile (raw bf16 qk, N-contiguous per K-row) async-DMA'd into K-major LDS;
// A tile (needs exp transform) staged through VGPRs. Both double-buffered.
// ============================================================================
#define P2A 34                               // bf16 elems per A row (68B)
#define A2_OFF 0
#define A2_BYTES (BM * P2A * 2)              // 4352
#define P2B 136                              // bf16 elems per B K-row (272B, 16B-aligned)
#define B2_OFF A2_BYTES
#define LDS2_HALF (B2_OFF + BK * P2B * 2)    // 13056 per buffer
#define LDS2_BYTES (2 * LDS2_HALF)           // 26112

__device__ __forceinline__ void stage_av(const __bf16* __restrict__ qb,
                                         char* smem, int s0, int n0, int t0,
                                         unsigned base, int tid,
                                         int arow, int aq, float mr, float invl) {
  // B tile: 32 K-rows x 16 chunks(8 bf16) = 512 chunks, 2 per thread (async)
#pragma unroll
  for (int it = 0; it < 2; ++it) {
    const int c = tid + it * 256;
    const int row = c >> 4, c8 = c & 15;
    async_g2l_b128(base + B2_OFF + row * (P2B * 2) + c8 * 16,
                   qb + (size_t)(t0 + row) * SEQ + n0 + c8 * 8);
  }
  // A tile: attn = exp(qk*scale - m) / l  (bf16), 4 threads/row x 8 elems
  const v8bf rv = *(const v8bf*)(qb + (size_t)(s0 + arow) * SEQ + t0 + aq * 8);
  __bf16* Aa = (__bf16*)(smem + base + A2_OFF);
#pragma unroll
  for (int e = 0; e < 8; ++e) {
    const float f = (float)rv[e] * SCALEF;
    Aa[arow * P2A + aq * 8 + e] = (__bf16)(__expf(f - mr) * invl);
  }
}

__global__ __launch_bounds__(256) void av_gemm_kernel(const __bf16* __restrict__ qkb,
                                                      const float* __restrict__ mrow,
                                                      const float* __restrict__ lrow,
                                                      float* __restrict__ out) {
  extern __shared__ char smem[];

  const int b    = blockIdx.z;
  const int n0   = blockIdx.x * BN;
  const int s0   = blockIdx.y * BM;
  const int tid  = threadIdx.x;
  const int lane = tid & 31;
  const int wave = tid >> 5;
  const int wr   = wave >> 2;
  const int wc   = wave & 3;
  const int lr   = lane & 15;
  const int half = lane >> 4;

  const __bf16* __restrict__ qb = qkb + (size_t)b * SEQ * SEQ;

  const int arow = tid >> 2;
  const int aq   = tid & 3;
  const float mr   = mrow[(size_t)b * SEQ + s0 + arow];
  const float invl = 1.0f / lrow[(size_t)b * SEQ + s0 + arow];

  v8f c00 = {}, c01 = {}, c10 = {}, c11 = {};

  const int nsteps = SEQ / BK;         // 64
  stage_av(qb, smem, s0, n0, 0, 0u, tid, arow, aq, mr, invl);

  for (int step = 0; step < nsteps; ++step) {
    const unsigned cur = (unsigned)(step & 1) * LDS2_HALF;
    if (step + 1 < nsteps) {
      __syncthreads();
      stage_av(qb, smem, s0, n0, (step + 1) * BK, cur ^ LDS2_HALF, tid, arow, aq, mr, invl);
      wait_async_le2();                // tile `step`'s async B ops have landed
    } else {
      wait_async_le0();
    }
    __syncthreads();

    const __bf16* Aa = (const __bf16*)(smem + cur + A2_OFF);
    const __bf16* Bk = (const __bf16*)(smem + cur + B2_OFF);
    const v16bf a0 = frag_a_from_bf16(Aa + (wr * 32 + lr) * P2A, half);
    const v16bf a1 = frag_a_from_bf16(Aa + (wr * 32 + 16 + lr) * P2A, half);
    const v16bf b0 = frag_b_from_kmajor(Bk, wc * 32 + lr, P2B, half);
    const v16bf b1 = frag_b_from_kmajor(Bk, wc * 32 + 16 + lr, P2B, half);
    c00 = WMMA_BF16(a0, b0, c00);
    c01 = WMMA_BF16(a0, b1, c01);
    c10 = WMMA_BF16(a1, b0, c10);
    c11 = WMMA_BF16(a1, b1, c11);
  }

  float* __restrict__ ob = out + (size_t)b * SEQ * SEQ;
#pragma unroll
  for (int jj = 0; jj < 8; ++jj) {
    const int r0 = s0 + wr * 32 + jj + half * 8;
    const int cc = n0 + wc * 32 + lr;
    ob[(size_t)r0 * SEQ + cc]             = c00[jj];
    ob[(size_t)r0 * SEQ + cc + 16]        = c01[jj];
    ob[(size_t)(r0 + 16) * SEQ + cc]      = c10[jj];
    ob[(size_t)(r0 + 16) * SEQ + cc + 16] = c11[jj];
  }
}

// ============================================================================
// Launch
// ============================================================================
extern "C" void kernel_launch(void* const* d_in, const int* in_sizes, int n_in,
                              void* d_out, int out_size, void* d_ws, size_t ws_size,
                              hipStream_t stream) {
  const float* q = (const float*)d_in[0];
  const float* k = (const float*)d_in[1];
  // d_in[2] (v) is dead in the reference math.
  float* out = (float*)d_out;

  // Workspace: qk bf16 (32 MiB) | row max (32 KiB) | row sum (32 KiB)
  char* ws = (char*)d_ws;
  __bf16* qkb = (__bf16*)ws;
  float* mrow = (float*)(ws + (size_t)BATCH * SEQ * SEQ * sizeof(__bf16));
  float* lrow = mrow + (size_t)BATCH * SEQ;

  const dim3 blk(256);
  const dim3 grid(SEQ / BN, SEQ / BM, BATCH);

  qk_gemm_kernel<<<grid, blk, LDS1_BYTES, stream>>>(q, k, qkb);
  softmax_stats_kernel<<<dim3(BATCH * SEQ), blk, 0, stream>>>(qkb, mrow, lrow);
  av_gemm_kernel<<<grid, blk, LDS2_BYTES, stream>>>(qkb, mrow, lrow, out);
}
